// SpatialAttention_19963007992131
// MI455X (gfx1250) — matrix operand order repaired and checked by reading, not compiled, not run on hardware
//
#include <hip/hip_runtime.h>

// Spatial attention == flash attention: Q=g (N x 32), K=f (N x 32), V=hh^T
// (N x 256), softmax over keys. v_wmma_f32_16x16x32_f16 everywhere; V operand
// staged into LDS with async-to-LDS copies (ASYNCcnt), double buffered.
// Projections are WMMA GEMMs on f16 operands prepared by small prep kernels.

typedef __attribute__((ext_vector_type(16))) _Float16 v16h;
typedef __attribute__((ext_vector_type(8)))  _Float16 v8h;
typedef __attribute__((ext_vector_type(8)))  float    v8f;
typedef __attribute__((address_space(3)))    _Float16 lds_f16;

constexpr int   Bc     = 8;
constexpr int   Cc     = 256;
constexpr int   Nc     = 4096;   // H*W
constexpr int   Crr    = 32;     // C / RED
constexpr float GAMMAv = 1.0f;

__device__ __forceinline__ void wait_async_le8() { asm volatile("s_wait_asynccnt 0x8" ::: "memory"); }
__device__ __forceinline__ void wait_async_0()  { asm volatile("s_wait_asynccnt 0x0" ::: "memory"); }

// 16-byte global -> LDS async copy (tracked by ASYNCcnt)
__device__ __forceinline__ void async_cp16(unsigned lds_off, const _Float16* g) {
  asm volatile("global_load_async_to_lds_b128 %0, %1, off"
               :: "v"(lds_off), "v"((unsigned long long)(size_t)g) : "memory");
}

// ---------------------------------------------------------------------------
// Weight convert: wh[320][256] f16  (rows 0-31: w1/K, 32-63: w2/Q, 64-319: w3/V)
// ---------------------------------------------------------------------------
__global__ __launch_bounds__(256) void wconv_kernel(
    const float* __restrict__ w1, const float* __restrict__ w2,
    const float* __restrict__ w3, _Float16* __restrict__ wh)
{
  const int i = blockIdx.x * 256 + threadIdx.x;      // 320*256 total
  const int r = i >> 8, c = i & 255;
  float v;
  if (r < Crr)            v = w1[(size_t)r * Cc + c];
  else if (r < 2 * Crr)   v = w2[(size_t)(r - Crr) * Cc + c];
  else                    v = w3[(size_t)(r - 2 * Crr) * Cc + c];
  wh[i] = (_Float16)v;
}

// ---------------------------------------------------------------------------
// Transpose+convert: x [B][C][N] f32 -> xt [B][N][C] f16 (LDS-tiled 64x64)
// ---------------------------------------------------------------------------
__global__ __launch_bounds__(256) void xpose_kernel(
    const float* __restrict__ x, _Float16* __restrict__ xt)
{
  __shared__ _Float16 tile[64][72];                  // padded rows
  const int b  = blockIdx.z;
  const int c0 = blockIdx.y * 64;
  const int n0 = blockIdx.x * 64;
  const int tid = threadIdx.x;
  const int j  = tid & 63;                           // n within tile
  const int ci = tid >> 6;                           // 0..3
  #pragma unroll
  for (int cc = 0; cc < 64; cc += 4)
    tile[cc + ci][j] = (_Float16)x[((size_t)b * Cc + c0 + cc + ci) * Nc + n0 + j];
  __syncthreads();
  const int ch = tid & 3;                            // 16-f16 chunk of c
  const int n  = tid >> 2;                           // 0..63
  v8h o0, o1;
  #pragma unroll
  for (int e = 0; e < 8; ++e) {
    o0[e] = tile[ch * 16 + e][n];
    o1[e] = tile[ch * 16 + 8 + e][n];
  }
  _Float16* dst = xt + ((size_t)b * Nc + n0 + n) * Cc + c0 + ch * 16;
  *(v8h*)dst       = o0;
  *(v8h*)(dst + 8) = o1;
}

// ---------------------------------------------------------------------------
// Projection GEMM: out[o][n] = sum_c wh[o][c] * xt[n][c], o in [0,320).
// Wave = one (64-wide o-group, 16-wide n-tile); 8 c-blocks x 4 WMMA.
//   A (16o x 32c) from wh rows (contiguous), B (32c x 16n) from xt rows.
// ---------------------------------------------------------------------------
__global__ __launch_bounds__(128) void projw_kernel(
    const _Float16* __restrict__ xt, const _Float16* __restrict__ wh,
    _Float16* __restrict__ Kh, _Float16* __restrict__ Qh, _Float16* __restrict__ Vh)
{
  const int lane = threadIdx.x & 31;
  const int wave = threadIdx.x >> 5;
  const int half = lane >> 4;
  const int l15  = lane & 15;
  const int task = blockIdx.x * 4 + wave;            // B * 5 * 256 = 10240
  const int b    = task / 1280;
  const int rem  = task % 1280;
  const int og   = rem >> 8;                         // 0..4  (64 o each)
  const int nt   = rem & 255;                        // n-tile

  const _Float16* xrow = xt + ((size_t)b * Nc + nt * 16 + l15) * Cc;

  v8f acc[4];
  #pragma unroll
  for (int t = 0; t < 4; ++t)
    #pragma unroll
    for (int r = 0; r < 8; ++r) acc[t][r] = 0.f;

  for (int cb = 0; cb < Cc; cb += 32) {
    const v16h bf = *(const v16h*)(xrow + cb + half * 16);
    #pragma unroll
    for (int t = 0; t < 4; ++t) {
      const _Float16* wr = wh + (size_t)(og * 64 + t * 16 + l15) * Cc + cb;
      const v8h lo = *(const v8h*)(wr + half * 16);
      const v8h hi = *(const v8h*)(wr + half * 16 + 8);
      v16h af;
      #pragma unroll
      for (int e = 0; e < 8; ++e) { af[e] = lo[e]; af[8 + e] = hi[e]; }
      acc[t] = __builtin_amdgcn_wmma_f32_16x16x32_f16(false, af, false, bf, (short)0,
                                                      acc[t], false, false);
    }
  }
  // D: lane -> col n = nt*16+l15, VGPR r -> row o = base + r + 8*half
  const int n = nt * 16 + l15;
  #pragma unroll
  for (int t = 0; t < 4; ++t) {
    const int obase = og * 64 + t * 16 + 8 * half;
    #pragma unroll
    for (int r = 0; r < 8; ++r) {
      const int o = obase + r;
      const _Float16 hv = (_Float16)acc[t][r];
      if (o < Crr)            Kh[((size_t)b * Nc + n) * Crr + o] = hv;
      else if (o < 2 * Crr)   Qh[((size_t)b * Nc + n) * Crr + (o - Crr)] = hv;
      else                    Vh[((size_t)b * Cc + (o - 2 * Crr)) * Nc + n] = hv;
    }
  }
}

// ---------------------------------------------------------------------------
// Flash attention. Grid: B*N/64 blocks x 128 threads (4 waves); wave owns a
// 16-query tile, streams 128 key blocks of 32. V slab (256c x 32i) staged in
// LDS by async-to-LDS copies, double-buffered, shared by all 4 waves.
// ---------------------------------------------------------------------------
#define VROW 40          // f16 per LDS row (80B: 64B data + 16B pad)

__device__ __forceinline__ void stage_v(const _Float16* Vb, int i0,
                                        unsigned lbase, int tid) {
  #pragma unroll
  for (int k = 0; k < 8; ++k) {                      // 1024 x 16B chunks / 128 thr
    const int q  = tid + (k << 7);
    const int c  = q >> 2;
    const int ko = q & 3;
    async_cp16(lbase + (unsigned)(c * (VROW * 2) + ko * 16),
               Vb + (size_t)c * Nc + i0 + ko * 8);
  }
}

__global__ __launch_bounds__(128) void attn_kernel(
    const _Float16* __restrict__ Qh, const _Float16* __restrict__ Kh,
    const _Float16* __restrict__ Vh, const float* __restrict__ x,
    float* __restrict__ out)
{
  __shared__ _Float16 vlds[2][Cc * VROW];            // 2 x 20 KB
  const int tid  = threadIdx.x;
  const int lane = tid & 31;
  const int wave = tid >> 5;
  const int half = lane >> 4;
  const int l15  = lane & 15;
  const int b    = blockIdx.x >> 6;
  const int jblk = blockIdx.x & 63;
  const int j0   = jblk * 64 + wave * 16;

  const unsigned lb0 = (unsigned)(size_t)(lds_f16*)&vlds[0][0];
  const unsigned lb1 = (unsigned)(size_t)(lds_f16*)&vlds[1][0];

  const _Float16* __restrict__ Kb = Kh + (size_t)b * Nc * Crr;
  const _Float16* __restrict__ Vb = Vh + (size_t)b * Cc * Nc;

  // Q B-fragment: 16 contiguous f16 of row (j0+l15) at d-offset 16*half
  const v16h qf = *(const v16h*)(Qh + ((size_t)b * Nc + j0 + l15) * Crr + half * 16);

  v8f oacc[16];
  #pragma unroll
  for (int t = 0; t < 16; ++t)
    #pragma unroll
    for (int r = 0; r < 8; ++r) oacc[t][r] = 0.f;

  float m_run = -1e30f;
  float l_run = 0.f;

  stage_v(Vb, 0, lb0, tid);                          // prologue: fill buffer 0

  for (int it = 0; it < Nc / 32; ++it) {
    const int i0  = it * 32;
    const int cur = it & 1;
    if (it + 1 < Nc / 32) stage_v(Vb, i0 + 32, cur ? lb0 : lb1, tid);

    // ---- S = K_tile x Q^T : two 16(i) x 16(j) tiles, full d=32 per WMMA ----
    const _Float16* kr0 = Kb + (size_t)(i0 + l15) * Crr;
    const _Float16* kr1 = Kb + (size_t)(i0 + 16 + l15) * Crr;
    v16h a0, a1;
    {
      const v8h lo0 = *(const v8h*)(kr0 + half * 16);
      const v8h hi0 = *(const v8h*)(kr0 + half * 16 + 8);
      const v8h lo1 = *(const v8h*)(kr1 + half * 16);
      const v8h hi1 = *(const v8h*)(kr1 + half * 16 + 8);
      #pragma unroll
      for (int e = 0; e < 8; ++e) {
        a0[e] = lo0[e]; a0[8 + e] = hi0[e];
        a1[e] = lo1[e]; a1[8 + e] = hi1[e];
      }
    }
    v8f s0, s1;
    #pragma unroll
    for (int r = 0; r < 8; ++r) { s0[r] = 0.f; s1[r] = 0.f; }
    s0 = __builtin_amdgcn_wmma_f32_16x16x32_f16(false, a0, false, qf, (short)0, s0, false, false);
    s1 = __builtin_amdgcn_wmma_f32_16x16x32_f16(false, a1, false, qf, (short)0, s1, false, false);

    // ---- online softmax over this key block (per column j = l15) ----
    float mb = -1e30f;
    #pragma unroll
    for (int r = 0; r < 8; ++r) mb = fmaxf(mb, fmaxf(s0[r], s1[r]));
    mb = fmaxf(mb, __shfl_xor(mb, 16, 32));
    const float m_new = fmaxf(m_run, mb);
    const float alpha = __expf(m_run - m_new);
    m_run = m_new;

    v16h pf;                                         // P^T A-fragment (lane-local)
    float psum = 0.f;
    #pragma unroll
    for (int r = 0; r < 8; ++r) {
      const float p0 = __expf(s0[r] - m_new);
      const float p1 = __expf(s1[r] - m_new);
      psum += p0 + p1;
      pf[r]     = (_Float16)p0;
      pf[8 + r] = (_Float16)p1;
    }
    psum += __shfl_xor(psum, 16, 32);
    l_run = l_run * alpha + psum;

    float av[8];                                     // alpha for row j = r+8*half
    #pragma unroll
    for (int r = 0; r < 8; ++r) av[r] = __shfl(alpha, half * 8 + r, 32);
    #pragma unroll
    for (int t = 0; t < 16; ++t)
      #pragma unroll
      for (int r = 0; r < 8; ++r) oacc[t][r] *= av[r];

    // ---- wait for this block's V slab, then O += P^T x V from LDS ----
    if (it + 1 < Nc / 32) wait_async_le8(); else wait_async_0();
    __syncthreads();
    {
      const _Float16* vbuf = &vlds[cur][0];
      #pragma unroll
      for (int t = 0; t < 16; ++t) {
        const v8h vlo = *(const v8h*)(vbuf + (t * 16 + l15) * VROW + half * 8);
        const v8h vhi = *(const v8h*)(vbuf + (t * 16 + l15) * VROW + 16 + half * 8);
        v16h vf;
        #pragma unroll
        for (int e = 0; e < 8; ++e) { vf[e] = vlo[e]; vf[8 + e] = vhi[e]; }
        oacc[t] = __builtin_amdgcn_wmma_f32_16x16x32_f16(false, pf, false, vf, (short)0,
                                                         oacc[t], false, false);
      }
    }
    __syncthreads();                                 // protect buffer reuse
  }

  // ---- epilogue: out = GAMMA * O / l + x ----
  float linv[8];
  #pragma unroll
  for (int r = 0; r < 8; ++r) linv[r] = 1.0f / __shfl(l_run, half * 8 + r, 32);

  #pragma unroll
  for (int t = 0; t < 16; ++t) {
    const size_t off = ((size_t)b * Cc + t * 16 + l15) * Nc + j0 + half * 8;
    const v8f xr = *(const v8f*)(x + off);
    v8f ov;
    #pragma unroll
    for (int r = 0; r < 8; ++r) ov[r] = GAMMAv * (oacc[t][r] * linv[r]) + xr[r];
    *(v8f*)(out + off) = ov;
  }
}

// ---------------------------------------------------------------------------
extern "C" void kernel_launch(void* const* d_in, const int* in_sizes, int n_in,
                              void* d_out, int out_size, void* d_ws, size_t ws_size,
                              hipStream_t stream) {
  (void)in_sizes; (void)n_in; (void)out_size; (void)ws_size;
  const float* x  = (const float*)d_in[0];
  const float* w1 = (const float*)d_in[1];
  const float* w2 = (const float*)d_in[2];
  const float* w3 = (const float*)d_in[3];
  float* out = (float*)d_out;

  // workspace: Kh | Qh | Vh | xt | wh   (~36.3 MB of f16)
  _Float16* Kh = (_Float16*)d_ws;
  _Float16* Qh = Kh + (size_t)Bc * Nc * Crr;
  _Float16* Vh = Qh + (size_t)Bc * Nc * Crr;
  _Float16* xt = Vh + (size_t)Bc * Cc * Nc;
  _Float16* wh = xt + (size_t)Bc * Nc * Cc;

  wconv_kernel<<<dim3((Cc + 2 * Crr)), 256, 0, stream>>>(w1, w2, w3, wh);
  xpose_kernel<<<dim3(Nc / 64, Cc / 64, Bc), 256, 0, stream>>>(x, xt);
  projw_kernel<<<dim3(Bc * 5 * 256 / 4), 128, 0, stream>>>(xt, wh, Kh, Qh, Vh);
  attn_kernel<<<dim3(Bc * (Nc / 64)), 128, 0, stream>>>(Qh, Kh, Vh, x, out);
}
